// SplitAttn_2765958938638
// MI455X (gfx1250) — matrix operand order repaired and checked by reading, not compiled, not run on hardware
//
#include <hip/hip_runtime.h>

#define DIMC 512
#define NHEAD 8
#define HD 64
#define NTOK 740
#define NTMP 256
#define NTGT 484
#define NPAD 768
#define BATCH 16
#define MTOT (BATCH*NPAD)
#define LT 16
#define LG 22
#define QSCALE 0.125f   /* 64^-0.5 folded into q projection */

typedef _Float16 v16h __attribute__((ext_vector_type(16)));
typedef _Float16 v8h  __attribute__((ext_vector_type(8)));
typedef float    v8f  __attribute__((ext_vector_type(8)));
typedef int      v4i  __attribute__((vector_size(16)));

union ABFrag { v16h v; v8h h[2]; };

__device__ __forceinline__ v8f wmma16(const ABFrag& a, const ABFrag& b, v8f c) {
  return __builtin_amdgcn_wmma_f32_16x16x32_f16(false, a.v, false, b.v, (short)0, c, false, false);
}
__device__ __forceinline__ v8f zero8() {
  v8f z = {0.f,0.f,0.f,0.f,0.f,0.f,0.f,0.f};
  return z;
}

/* gfx1250 async global->LDS path (device pass only; host pass parses fallback) */
#if defined(__gfx1250__) && __has_builtin(__builtin_amdgcn_global_load_async_to_lds_b128)
#define HAVE_ASYNC 1
#else
#define HAVE_ASYNC 0
#endif

#if HAVE_ASYNC
__device__ __forceinline__ void async_cp16(_Float16* lds, const _Float16* g) {
  __builtin_amdgcn_global_load_async_to_lds_b128(
      (__attribute__((address_space(1))) v4i*)(g),
      (__attribute__((address_space(3))) v4i*)(lds), 0, 0);
}
template<int N>
__device__ __forceinline__ void wait_async() {
#if __has_builtin(__builtin_amdgcn_s_wait_asynccnt)
  __builtin_amdgcn_s_wait_asynccnt((short)N);
#else
  asm volatile("s_wait_asynccnt %0" :: "i"(N) : "memory");
#endif
}
#endif

/* ---------------- weight transpose + f16 convert: wT[n][k] = w[k][n] -------- */
__global__ __launch_bounds__(256) void k_wt(
    const float* __restrict__ w0, const float* __restrict__ w1,
    const float* __restrict__ w2, const float* __restrict__ w3,
    _Float16* __restrict__ t0, _Float16* __restrict__ t1,
    _Float16* __restrict__ t2, _Float16* __restrict__ t3)
{
  int t = blockIdx.x*256 + threadIdx.x;          // t = n*512 + k
  int n = t >> 9, k2 = t & (DIMC-1);
  int src = k2*DIMC + n;
  t0[t] = (_Float16)w0[src];
  t1[t] = (_Float16)w1[src];
  t2[t] = (_Float16)w2[src];
  t3[t] = (_Float16)w3[src];
}

/* ---------------- gathered relative-position bias tables -------------------- */
__global__ __launch_bounds__(256) void k_bias(
    const float* __restrict__ rpbT, const float* __restrict__ rpbM,
    const float* __restrict__ tt,   const float* __restrict__ tg,
    float* __restrict__ rel, float* __restrict__ tb)
{
  int t = blockIdx.x*256 + threadIdx.x;
  const int relCount = NHEAD*NTGT*NTOK;
  if (t < relCount) {
    int h  = t / (NTGT*NTOK);
    int r  = t - h*(NTGT*NTOK);
    int q  = r / NTOK;
    int kk = r - q*NTOK;
    float v;
    if (kk < NTMP) {
      v = tt[h*NTMP + kk] + tg[h*NTGT + q];
    } else {
      int ks = kk - NTMP;
      int qi = q/LG,  qj = q - qi*LG;
      int ki = ks/LG, kj = ks - ki*LG;
      int idx = (qi-ki+LG-1)*(2*LG-1) + (qj-kj+LG-1);
      v = rpbT[idx*NHEAD + h];
    }
    rel[t] = v;
  } else {
    int t2 = t - relCount;
    if (t2 < NTMP*NTMP) {
      int q = t2 >> 8, kk = t2 & 255;
      int qi = q>>4,  qj = q&15;
      int ki = kk>>4, kj = kk&15;
      int idx = (qi-ki+LT-1)*(2*LT-1) + (qj-kj+LT-1);
      tb[t2] = rpbM[idx];
    }
  }
}

/* ---------------- depthwise 3x3 conv + BN, 3 branches, f16 out -------------- */
__global__ __launch_bounds__(256) void k_conv_bn(
    const float* __restrict__ x,
    const float* __restrict__ wq, const float* __restrict__ gq, const float* __restrict__ bq,
    const float* __restrict__ mq, const float* __restrict__ vq,
    const float* __restrict__ wk, const float* __restrict__ gk, const float* __restrict__ bk,
    const float* __restrict__ mk, const float* __restrict__ vk,
    const float* __restrict__ wv, const float* __restrict__ gv, const float* __restrict__ bv,
    const float* __restrict__ mv, const float* __restrict__ vv,
    _Float16* __restrict__ oq, _Float16* __restrict__ ok, _Float16* __restrict__ ov)
{
  long gid = (long)blockIdx.x*256 + threadIdx.x;   // m*512 + c, m in [0,MTOT)
  int c = (int)(gid & (DIMC-1));
  int m = (int)(gid >> 9);
  int b = m / NPAD, n = m - b*NPAD;
  long o = gid;
  if (n >= NTOK) {                                  // zero the pad rows
    oq[o] = (_Float16)0.f; ok[o] = (_Float16)0.f; ov[o] = (_Float16)0.f;
    return;
  }
  int L, base, p;
  if (n < NTMP) { L = LT; base = 0;    p = n; }
  else          { L = LG; base = NTMP; p = n - NTMP; }
  int i = p / L, j = p - i*L;
  float sq = 0.f, sk = 0.f, sv = 0.f;
  #pragma unroll
  for (int ky = 0; ky < 3; ky++) {
    #pragma unroll
    for (int kx = 0; kx < 3; kx++) {
      int ii = i + ky - 1, jj = j + kx - 1;
      if (ii >= 0 && ii < L && jj >= 0 && jj < L) {
        float xv = x[((long)b*NTOK + base + ii*L + jj)*DIMC + c];
        int wo = c*9 + ky*3 + kx;
        sq += xv*wq[wo]; sk += xv*wk[wo]; sv += xv*wv[wo];
      }
    }
  }
  float iq = gq[c]*rsqrtf(vq[c] + 1e-5f);
  float ik = gk[c]*rsqrtf(vk[c] + 1e-5f);
  float iv = gv[c]*rsqrtf(vv[c] + 1e-5f);
  oq[o] = (_Float16)((sq - mq[c])*iq + bq[c]);
  ok[o] = (_Float16)((sk - mk[c])*ik + bk[c]);
  ov[o] = (_Float16)((sv - mv[c])*iv + bv[c]);
}

/* ---------------- WMMA GEMM: Y[M,512] = X[M,512] @ WT^T + bias -------------- */
/* Block = 64(M) x 128(N), 8 waves of 32x32.  Device build stages tiles in LDS
   via async global->LDS b128 copies (ASYNCcnt double buffer); otherwise direct
   global-load fragments.  mode 0: f16 out (stride 512); mode 1: f32 out with
   NPAD->NTOK row compaction into d_out.                                        */
__global__ __launch_bounds__(256) void k_gemm(
    const _Float16* __restrict__ X, const _Float16* __restrict__ WT,
    const float* __restrict__ bias, void* __restrict__ outp,
    float outScale, int mode)
{
  int tid  = threadIdx.x;
  int lane = tid & 31, wvid = tid >> 5;
  int ll = lane & 15, lh = lane >> 4;
  int wm = wvid >> 2, wn = wvid & 3;
  int mBlk = blockIdx.x*64, nBlk = blockIdx.y*128;
  int m0 = mBlk + wm*32;
  int n0 = nBlk + wn*32;
  v8f c[2][2];
  c[0][0] = zero8(); c[0][1] = zero8(); c[1][0] = zero8(); c[1][1] = zero8();

#if HAVE_ASYNC
  __shared__ __align__(16) _Float16 sA[2][64*32];
  __shared__ __align__(16) _Float16 sB[2][128*32];
  int sRow   = tid >> 2;           // 0..63
  int sChunk = (tid & 3) * 8;      // half offset 0/8/16/24 within a 32-half row
  const _Float16* gA  = X  + (long)(mBlk + sRow)*DIMC + sChunk;
  const _Float16* gB0 = WT + (long)(nBlk + sRow)*DIMC + sChunk;
  const _Float16* gB1 = WT + (long)(nBlk + 64 + sRow)*DIMC + sChunk;

  /* stage 0 */
  async_cp16(&sA[0][tid*8],        gA);
  async_cp16(&sB[0][tid*8],        gB0);
  async_cp16(&sB[0][2048 + tid*8], gB1);

  for (int t = 0; t < 16; t++) {
    int cur = t & 1;
    if (t + 1 < 16) {
      int k1 = (t + 1) * 32;
      async_cp16(&sA[cur^1][tid*8],        gA  + k1);
      async_cp16(&sB[cur^1][tid*8],        gB0 + k1);
      async_cp16(&sB[cur^1][2048 + tid*8], gB1 + k1);
      wait_async<3>();               // stage t complete, stage t+1 in flight
    } else {
      wait_async<0>();
    }
    __syncthreads();                 // tile visible to all waves
    const _Float16* aBase = &sA[cur][0];
    const _Float16* bBase = &sB[cur][0];
    ABFrag a[2], bf[2];
    #pragma unroll
    for (int i = 0; i < 2; i++) {
      const _Float16* ar = aBase + (wm*32 + i*16 + ll)*32;
      const _Float16* br = bBase + (wn*32 + i*16 + ll)*32;
      a[i].h[0]  = *(const v8h*)(ar + lh*8);
      a[i].h[1]  = *(const v8h*)(ar + 16 + lh*8);
      bf[i].h[0] = *(const v8h*)(br + lh*8);
      bf[i].h[1] = *(const v8h*)(br + 16 + lh*8);
    }
    #pragma unroll
    for (int i = 0; i < 2; i++)
      #pragma unroll
      for (int j = 0; j < 2; j++)
        c[i][j] = wmma16(a[i], bf[j], c[i][j]);
    __syncthreads();                 // done reading before buffer is re-filled
  }
#else
  const _Float16* xr[2] = { X + (long)(m0 + ll)*DIMC, X + (long)(m0 + 16 + ll)*DIMC };
  const _Float16* wr[2] = { WT + (long)(n0 + ll)*DIMC, WT + (long)(n0 + 16 + ll)*DIMC };
  for (int k0 = 0; k0 < DIMC; k0 += 32) {
    ABFrag a[2], bf[2];
    #pragma unroll
    for (int i = 0; i < 2; i++) {
      a[i].h[0]  = *(const v8h*)(xr[i] + k0 + lh*8);
      a[i].h[1]  = *(const v8h*)(xr[i] + k0 + 16 + lh*8);
      bf[i].h[0] = *(const v8h*)(wr[i] + k0 + lh*8);
      bf[i].h[1] = *(const v8h*)(wr[i] + k0 + 16 + lh*8);
    }
    #pragma unroll
    for (int i = 0; i < 2; i++)
      #pragma unroll
      for (int j = 0; j < 2; j++)
        c[i][j] = wmma16(a[i], bf[j], c[i][j]);
  }
#endif

  #pragma unroll
  for (int i = 0; i < 2; i++) {
    #pragma unroll
    for (int j = 0; j < 2; j++) {
      #pragma unroll
      for (int r = 0; r < 8; r++) {
        int row = m0 + i*16 + r + lh*8;
        int col = n0 + j*16 + ll;
        float val = (c[i][j][r] + bias[col]) * outScale;
        if (mode == 0) {
          ((_Float16*)outp)[(long)row*DIMC + col] = (_Float16)val;
        } else {
          int bb = row / NPAD, nn = row - bb*NPAD;
          if (nn < NTOK)
            ((float*)outp)[((long)bb*NTOK + nn)*DIMC + col] = val;
        }
      }
    }
  }
}

/* ---------------- V transpose: vT[b][c][key] = v[b*NPAD+key][c] ------------- */
__global__ __launch_bounds__(256) void k_vt(
    const _Float16* __restrict__ v, _Float16* __restrict__ vt)
{
  __shared__ _Float16 tile[16][17];
  int tx = threadIdx.x, ty = threadIdx.y;
  int m0 = blockIdx.x*16, c0 = blockIdx.y*16;
  tile[ty][tx] = v[(long)(m0 + ty)*DIMC + c0 + tx];
  __syncthreads();
  int m = m0 + tx;
  int b = m / NPAD, n = m - b*NPAD;
  vt[(long)b*DIMC*NPAD + (long)(c0 + ty)*NPAD + n] = tile[tx][ty];
}

/* ---------------- flash attention, one wave per (b, h, 16-query tile) ------- */
__global__ __launch_bounds__(256) void k_attn(
    const _Float16* __restrict__ Q,  const _Float16* __restrict__ Kp,
    const _Float16* __restrict__ VT, const float* __restrict__ rel,
    const float* __restrict__ tb,    _Float16* __restrict__ O)
{
  __shared__ __align__(16) _Float16 pS[8][16*32];
  int lane = threadIdx.x & 31, wid = threadIdx.x >> 5;
  int ll = lane & 15, lh = lane >> 4;
  int unit = blockIdx.x*8 + wid;
  const int TUNITS = BATCH*NHEAD*31;                     // 3968 target units
  int b, h, qt, nKeys, nChunks, nQ, biasStride, qRowBase;
  const float* biasBase;
  if (unit < TUNITS) {
    b = unit / (NHEAD*31);
    int r = unit - b*(NHEAD*31);
    h = r / 31; qt = r - h*31;
    qRowBase = NTMP + qt*16;
    nKeys = NTOK; nChunks = 24; nQ = NTGT;
    biasBase = rel + (long)h*NTGT*NTOK; biasStride = NTOK;
  } else {
    int u = unit - TUNITS;
    b = u / (NHEAD*16);
    int r = u - b*(NHEAD*16);
    h = r / 16; qt = r - h*16;
    qRowBase = qt*16;
    nKeys = NTMP; nChunks = 8; nQ = NTMP;
    biasBase = tb; biasStride = NTMP;
  }
  int qBiasBase = qt*16;

  ABFrag qa[2];
  const _Float16* qrow = Q + ((long)(b*NPAD + qRowBase + ll))*DIMC + h*HD;
  #pragma unroll
  for (int cc = 0; cc < 2; cc++) {
    qa[cc].h[0] = *(const v8h*)(qrow + cc*32 + lh*8);
    qa[cc].h[1] = *(const v8h*)(qrow + cc*32 + 16 + lh*8);
  }

  v8f acc[4];
  #pragma unroll
  for (int j = 0; j < 4; j++) acc[j] = zero8();
  float mr[8], lr[8];
  #pragma unroll
  for (int r = 0; r < 8; r++) { mr[r] = -1e30f; lr[r] = 0.f; }

  _Float16* pb = pS[wid];
  const _Float16* kBase = Kp + (long)b*NPAD*DIMC + h*HD;
  const _Float16* vBase = VT + (long)b*DIMC*NPAD + (long)h*HD*NPAD;

  for (int t = 0; t < nChunks; t++) {
    int k0 = t*32;
    v8f s[2];
    #pragma unroll
    for (int sj = 0; sj < 2; sj++) {
      int key = k0 + sj*16 + ll;
      const _Float16* kr = kBase + (long)key*DIMC;
      v8f sa = zero8();
      #pragma unroll
      for (int cc = 0; cc < 2; cc++) {
        ABFrag kb;
        kb.h[0] = *(const v8h*)(kr + cc*32 + lh*8);
        kb.h[1] = *(const v8h*)(kr + cc*32 + 16 + lh*8);
        sa = wmma16(qa[cc], kb, sa);
      }
      s[sj] = sa;
    }
    /* bias + online softmax (rows live in 16-lane halves) */
    #pragma unroll
    for (int r = 0; r < 8; r++) {
      int qq  = qBiasBase + r + lh*8;
      int qcl = qq < nQ ? qq : nQ - 1;
      float sv0, sv1;
      {
        int key = k0 + ll;
        float bias = -1e30f;
        if (key < nKeys) bias = biasBase[(long)qcl*biasStride + key];
        sv0 = s[0][r] + bias;
      }
      {
        int key = k0 + 16 + ll;
        float bias = -1e30f;
        if (key < nKeys) bias = biasBase[(long)qcl*biasStride + key];
        sv1 = s[1][r] + bias;
      }
      float rm = fmaxf(sv0, sv1);
      rm = fmaxf(rm, __shfl_xor(rm, 1));
      rm = fmaxf(rm, __shfl_xor(rm, 2));
      rm = fmaxf(rm, __shfl_xor(rm, 4));
      rm = fmaxf(rm, __shfl_xor(rm, 8));
      float mnew = fmaxf(mr[r], rm);
      float sc = __expf(mr[r] - mnew);
      mr[r] = mnew;
      float p0 = __expf(sv0 - mnew);
      float p1 = __expf(sv1 - mnew);
      float ps = p0 + p1;
      ps += __shfl_xor(ps, 1);
      ps += __shfl_xor(ps, 2);
      ps += __shfl_xor(ps, 4);
      ps += __shfl_xor(ps, 8);
      lr[r] = lr[r]*sc + ps;
      #pragma unroll
      for (int j = 0; j < 4; j++) acc[j][r] *= sc;
      int prow = r + lh*8;
      pb[prow*32 + ll]      = (_Float16)p0;
      pb[prow*32 + 16 + ll] = (_Float16)p1;
    }
    /* wave-local LDS fence: P written cross-lane, reload as A-fragment */
    asm volatile("s_wait_dscnt 0x0" ::: "memory");
    ABFrag pa;
    pa.h[0] = *(const v8h*)(pb + ll*32 + lh*8);
    pa.h[1] = *(const v8h*)(pb + ll*32 + 16 + lh*8);
    #pragma unroll
    for (int j = 0; j < 4; j++) {
      const _Float16* vr = vBase + (long)(j*16 + ll)*NPAD + k0;
      ABFrag vb;
      vb.h[0] = *(const v8h*)(vr + lh*8);
      vb.h[1] = *(const v8h*)(vr + 16 + lh*8);
      acc[j] = wmma16(pa, vb, acc[j]);
    }
  }
  /* normalize + store */
  #pragma unroll
  for (int r = 0; r < 8; r++) {
    float inv = 1.0f / lr[r];
    int rr = r + lh*8;
    int qq = qBiasBase + rr;
    bool okrow = qq < nQ;
    long row = (long)(b*NPAD + qRowBase + rr);
    #pragma unroll
    for (int j = 0; j < 4; j++) {
      if (okrow) O[row*DIMC + h*HD + j*16 + ll] = (_Float16)(acc[j][r]*inv);
    }
  }
}

/* ---------------------------------------------------------------------------- */
extern "C" void kernel_launch(void* const* d_in, const int* in_sizes, int n_in,
                              void* d_out, int out_size, void* d_ws, size_t ws_size,
                              hipStream_t stream)
{
  const float* x    = (const float*)d_in[0];
  const float* cqw  = (const float*)d_in[1];
  const float* gq   = (const float*)d_in[2];
  const float* bq_  = (const float*)d_in[3];
  const float* mq   = (const float*)d_in[4];
  const float* vq   = (const float*)d_in[5];
  const float* ckw  = (const float*)d_in[6];
  const float* gk   = (const float*)d_in[7];
  const float* bk_  = (const float*)d_in[8];
  const float* mk   = (const float*)d_in[9];
  const float* vk   = (const float*)d_in[10];
  const float* cvw  = (const float*)d_in[11];
  const float* gv   = (const float*)d_in[12];
  const float* bv_  = (const float*)d_in[13];
  const float* mv   = (const float*)d_in[14];
  const float* vv   = (const float*)d_in[15];
  const float* wq   = (const float*)d_in[16];
  const float* bq   = (const float*)d_in[17];
  const float* wk   = (const float*)d_in[18];
  const float* bk   = (const float*)d_in[19];
  const float* wv   = (const float*)d_in[20];
  const float* bv   = (const float*)d_in[21];
  const float* wp   = (const float*)d_in[22];
  const float* bp   = (const float*)d_in[23];
  const float* rpbT = (const float*)d_in[24];
  const float* rpbM = (const float*)d_in[25];
  const float* tt   = (const float*)d_in[26];
  const float* tg   = (const float*)d_in[27];

  char* ws = (char*)d_ws;
  const size_t S1  = (size_t)MTOT*DIMC*2;        // one f16 [MTOT,512] buffer
  const size_t WTS = (size_t)DIMC*DIMC*2;        // one f16 weight matrix
  _Float16* cq  = (_Float16*)(ws + 0*S1);
  _Float16* ck  = (_Float16*)(ws + 1*S1);
  _Float16* cv  = (_Float16*)(ws + 2*S1);
  _Float16* qb  = (_Float16*)(ws + 3*S1);
  _Float16* kb  = (_Float16*)(ws + 4*S1);
  _Float16* vb  = (_Float16*)(ws + 5*S1);
  _Float16* ao  = (_Float16*)(ws + 6*S1);
  _Float16* wqT = (_Float16*)(ws + 7*S1);
  _Float16* wkT = (_Float16*)(ws + 7*S1 + WTS);
  _Float16* wvT = (_Float16*)(ws + 7*S1 + 2*WTS);
  _Float16* wpT = (_Float16*)(ws + 7*S1 + 3*WTS);
  float*    rel = (float*)   (ws + 7*S1 + 4*WTS);
  float*    tb  = (float*)   (ws + 7*S1 + 4*WTS + (size_t)NHEAD*NTGT*NTOK*4);
  _Float16* vT  = cq;   // alias: conv_q buffer is dead once the q-GEMM has run

  k_wt<<<DIMC*DIMC/256, 256, 0, stream>>>(wq, wk, wv, wp, wqT, wkT, wvT, wpT);
  int biasTot = NHEAD*NTGT*NTOK + NTMP*NTMP;
  k_bias<<<(biasTot + 255)/256, 256, 0, stream>>>(rpbT, rpbM, tt, tg, rel, tb);
  k_conv_bn<<<MTOT*DIMC/256, 256, 0, stream>>>(x,
      cqw, gq, bq_, mq, vq,
      ckw, gk, bk_, mk, vk,
      cvw, gv, bv_, mv, vv,
      cq, ck, cv);

  dim3 gg(MTOT/64, DIMC/128);
  k_gemm<<<gg, 256, 0, stream>>>(cq, wqT, bq, (void*)qb, QSCALE, 0);
  k_gemm<<<gg, 256, 0, stream>>>(ck, wkT, bk, (void*)kb, 1.0f,   0);
  k_gemm<<<gg, 256, 0, stream>>>(cv, wvT, bv, (void*)vb, 1.0f,   0);

  k_vt<<<dim3(MTOT/16, DIMC/16), dim3(16,16), 0, stream>>>(vb, vT);

  int units = BATCH*NHEAD*31 + BATCH*NHEAD*16;   // 3968 + 2048 = 6016
  k_attn<<<units/8, 256, 0, stream>>>(qb, kb, vT, rel, tb, ao);

  k_gemm<<<gg, 256, 0, stream>>>(ao, wpT, bp, d_out, 1.0f, 1);
}
